// LabelCrossAttentionDE_51642686767876
// MI455X (gfx1250) — matrix operand order repaired and checked by reading, not compiled, not run on hardware
//
#include <hip/hip_runtime.h>
#include <hip/hip_bf16.h>

#define D_ 768
#define C_ 8000
#define S_ 4096
#define B_ 2
#define LN_EPS 1e-5f

typedef __attribute__((ext_vector_type(16))) __bf16       v16bf;
typedef __attribute__((ext_vector_type(8)))  float        v8f;
typedef __attribute__((ext_vector_type(4)))  unsigned int u32x4;

union FragB { v16bf v; u32x4 q[2]; };
union Pack8 { unsigned short s[8]; u32x4 q; };

__device__ __forceinline__ unsigned short f2bf(float f) {
  union { float f; unsigned u; } x; x.f = f;
  unsigned r = x.u + 0x7FFFu + ((x.u >> 16) & 1u);   // round-to-nearest-even
  return (unsigned short)(r >> 16);
}

// ---------------------------------------------------------------- convert ----
__global__ void cvt_f32_bf16(const float* __restrict__ in,
                             unsigned short* __restrict__ out, int n) {
  int i = blockIdx.x * blockDim.x + threadIdx.x;
  if (i < n) out[i] = f2bf(in[i]);
}

// ------------------------------------------------------------- K/V GEMMs ----
// blockIdx.x: s-tile (S/16), blockIdx.y: group of 8 e-tiles, blockIdx.z: b*2+isV
// Each wave computes one 16x16 output tile with a K=768 loop of bf16 WMMA.
// K is stored row-major K[b][s][e]; V is stored transposed Vt[b][e][s] (+bias).
__global__ __launch_bounds__(256)
void kv_gemm(const unsigned short* __restrict__ xbf,
             const unsigned short* __restrict__ wkbf,
             const unsigned short* __restrict__ wvbf,
             const float* __restrict__ bv,
             unsigned short* __restrict__ kbf,
             unsigned short* __restrict__ vtbf) {
  const int w  = threadIdx.x >> 5;
  const int L  = threadIdx.x & 31;
  const int h  = L >> 4;      // half-wave
  const int ln = L & 15;      // lane-in-half
  const int s0 = blockIdx.x * 16;
  const int e0 = (blockIdx.y * 8 + w) * 16;
  const int b  = blockIdx.z >> 1;
  const int isV = blockIdx.z & 1;

  v8f acc = {};
  if (!isV) {
    // D = Wk (M=e,K=d) x x^T-as-B (K=d,N=s)  -> D[e][s] = K[s][e]
    const unsigned short* arow = wkbf + (size_t)(e0 + ln) * D_;
    const unsigned short* brow = xbf + ((size_t)b * S_ + s0 + ln) * D_;
    #pragma unroll 4
    for (int kk = 0; kk < D_; kk += 32) {
      FragB a, bb;
      a.q[0]  = *(const u32x4*)(arow + kk + 8 * h);
      a.q[1]  = *(const u32x4*)(arow + kk + 8 * h + 16);
      bb.q[0] = *(const u32x4*)(brow + kk + 8 * h);
      bb.q[1] = *(const u32x4*)(brow + kk + 8 * h + 16);
      acc = __builtin_amdgcn_wmma_f32_16x16x32_bf16(false, a.v, false, bb.v,
                                                    (short)0, acc, false, false);
    }
    // lane holds column s = s0+ln, rows e = e0 + j + 8h -> contiguous-e b128 store
    Pack8 pk;
    #pragma unroll
    for (int j = 0; j < 8; ++j) pk.s[j] = f2bf(acc[j]);
    *(u32x4*)(kbf + ((size_t)b * S_ + s0 + ln) * D_ + e0 + 8 * h) = pk.q;
  } else {
    // D = x (M=s,K=d) x Wv^T-as-B (K=d,N=e) -> D[s][e] = V[s][e]
    const unsigned short* arow = xbf + ((size_t)b * S_ + s0 + ln) * D_;
    const unsigned short* brow = wvbf + (size_t)(e0 + ln) * D_;
    #pragma unroll 4
    for (int kk = 0; kk < D_; kk += 32) {
      FragB a, bb;
      a.q[0]  = *(const u32x4*)(arow + kk + 8 * h);
      a.q[1]  = *(const u32x4*)(arow + kk + 8 * h + 16);
      bb.q[0] = *(const u32x4*)(brow + kk + 8 * h);
      bb.q[1] = *(const u32x4*)(brow + kk + 8 * h + 16);
      acc = __builtin_amdgcn_wmma_f32_16x16x32_bf16(false, a.v, false, bb.v,
                                                    (short)0, acc, false, false);
    }
    const float bias = bv[e0 + ln];  // lane column e is fixed
    Pack8 pk;
    #pragma unroll
    for (int j = 0; j < 8; ++j) pk.s[j] = f2bf(acc[j] + bias);
    // rows s = s0 + j + 8h contiguous -> b128 store into Vt[b][e][s]
    *(u32x4*)(vtbf + ((size_t)b * D_ + e0 + ln) * S_ + s0 + 8 * h) = pk.q;
  }
}

// --------------------------------------------------- fused flash attention ----
// Grid (C/32, B), block 256 = 8 waves = 2 label-tiles x 4 d-slices(192 each).
__global__ __launch_bounds__(256)
void attn(const unsigned short* __restrict__ qbf,
          const unsigned short* __restrict__ kbf,
          const unsigned short* __restrict__ vtbf,
          const float* __restrict__ gamma,
          const float* __restrict__ beta,
          const float* __restrict__ out_w,
          const float* __restrict__ out_b,
          float* __restrict__ out) {
  __shared__ float          lp[2][4][16][36];   // per-slice partial logits
  __shared__ unsigned short Pl[2][16][40];      // softmaxed P tile (bf16, padded)
  __shared__ float          alphaL[2][16];
  __shared__ float          lL[2][16];
  __shared__ float          rsL[2][16], rssL[2][16], orL[2][16];

  const int tid = threadIdx.x;
  const int w = tid >> 5, L = tid & 31, h = L >> 4, ln = L & 15;
  const int ct = w >> 2;        // label-tile within WG
  const int p  = w & 3;         // d-slice
  const int c0 = blockIdx.x * 32 + ct * 16;
  const int b  = blockIdx.y;
  const int dbase = p * 192;

  if (tid < 32) {
    (&rsL[0][0])[tid]  = 0.f;
    (&rssL[0][0])[tid] = 0.f;
    (&orL[0][0])[tid]  = 0.f;
  }

  // Q A-fragments resident in registers for the whole S loop (48 VGPRs)
  FragB qf[6];
  {
    const unsigned short* qrow = qbf + (size_t)(c0 + ln) * D_ + dbase;
    #pragma unroll
    for (int kk = 0; kk < 6; ++kk) {
      qf[kk].q[0] = *(const u32x4*)(qrow + kk * 32 + 8 * h);
      qf[kk].q[1] = *(const u32x4*)(qrow + kk * 32 + 8 * h + 16);
    }
  }

  v8f acc[12] = {};                 // y accumulators: 16 rows x 192 cols, f32
  float m_r = -1e30f, l_r = 0.f;    // online softmax state (leader waves)

  for (int it = 0; it < S_ / 32; ++it) {
    const int s0 = it * 32;
    // ---- partial logits over this wave's 192-wide d slice (6 WMMA / s-tile)
    #pragma unroll
    for (int st = 0; st < 2; ++st) {
      v8f al = {};
      const unsigned short* krow =
          kbf + ((size_t)b * S_ + s0 + st * 16 + ln) * D_ + dbase;
      #pragma unroll
      for (int kk = 0; kk < 6; ++kk) {
        FragB kb;
        kb.q[0] = *(const u32x4*)(krow + kk * 32 + 8 * h);
        kb.q[1] = *(const u32x4*)(krow + kk * 32 + 8 * h + 16);
        al = __builtin_amdgcn_wmma_f32_16x16x32_bf16(false, qf[kk].v, false, kb.v,
                                                     (short)0, al, false, false);
      }
      #pragma unroll
      for (int j = 0; j < 8; ++j)
        lp[ct][p][j + 8 * h][st * 16 + ln] = al[j];
    }
    if (it + 1 < S_ / 32)   // warm L2->L0 for the next K chunk
      __builtin_prefetch(kbf + ((size_t)b * S_ + s0 + 32 + ln) * D_ + dbase, 0, 1);
    __syncthreads();

    // ---- online softmax: one wave per label-tile, lane = row, half = s-half
    if (p == 0) {
      float vals[16];
      float rmax = -1e30f;
      #pragma unroll
      for (int c2 = 0; c2 < 16; ++c2) {
        const int col = 16 * h + c2;
        float v = lp[ct][0][ln][col] + lp[ct][1][ln][col] +
                  lp[ct][2][ln][col] + lp[ct][3][ln][col];
        vals[c2] = v;
        rmax = fmaxf(rmax, v);
      }
      rmax = fmaxf(rmax, __shfl_xor(rmax, 16, 32));
      const float mnew  = fmaxf(m_r, rmax);
      const float alpha = __expf(m_r - mnew);
      float ps = 0.f;
      union { unsigned short s[16]; u32x4 q[2]; } pk;
      #pragma unroll
      for (int c2 = 0; c2 < 16; ++c2) {
        const float pv = __expf(vals[c2] - mnew);
        ps += pv;
        pk.s[c2] = f2bf(pv);
      }
      ps += __shfl_xor(ps, 16, 32);
      l_r = l_r * alpha + ps;
      m_r = mnew;
      alphaL[ct][ln] = alpha;
      *(u32x4*)&Pl[ct][ln][16 * h]     = pk.q[0];
      *(u32x4*)&Pl[ct][ln][16 * h + 8] = pk.q[1];
    }
    __syncthreads();

    // ---- rescale accumulators and acc += P x V  (12 WMMA per wave)
    float av[8];
    #pragma unroll
    for (int j = 0; j < 8; ++j) av[j] = alphaL[ct][j + 8 * h];
    FragB pf;
    pf.q[0] = *(const u32x4*)&Pl[ct][ln][8 * h];
    pf.q[1] = *(const u32x4*)&Pl[ct][ln][8 * h + 16];
    #pragma unroll
    for (int t = 0; t < 12; ++t) {
      #pragma unroll
      for (int j = 0; j < 8; ++j) acc[t][j] *= av[j];
      FragB vb;
      const unsigned short* vrow =
          vtbf + ((size_t)b * D_ + dbase + t * 16 + ln) * S_ + s0;
      vb.q[0] = *(const u32x4*)(vrow + 8 * h);
      vb.q[1] = *(const u32x4*)(vrow + 8 * h + 16);
      acc[t] = __builtin_amdgcn_wmma_f32_16x16x32_bf16(false, pf.v, false, vb.v,
                                                       (short)0, acc[t], false, false);
    }
  }

  // ---------------- epilogue: 1/l, LayerNorm stats, out_w contraction --------
  if (p == 0) lL[ct][ln] = l_r;     // both halves hold identical values
  __syncthreads();

  float li[8];
  #pragma unroll
  for (int j = 0; j < 8; ++j) li[j] = 1.0f / lL[ct][j + 8 * h];

  float s1[8], s2[8];
  #pragma unroll
  for (int j = 0; j < 8; ++j) { s1[j] = 0.f; s2[j] = 0.f; }
  #pragma unroll
  for (int t = 0; t < 12; ++t)
    #pragma unroll
    for (int j = 0; j < 8; ++j) {
      const float y = acc[t][j] * li[j];
      acc[t][j] = y;
      s1[j] += y;
      s2[j] += y * y;
    }
  #pragma unroll
  for (int j = 0; j < 8; ++j) {
    #pragma unroll
    for (int msk = 1; msk <= 8; msk <<= 1) {
      s1[j] += __shfl_xor(s1[j], msk, 32);
      s2[j] += __shfl_xor(s2[j], msk, 32);
    }
  }
  if (ln == 0) {
    #pragma unroll
    for (int j = 0; j < 8; ++j) {
      atomicAdd(&rsL[ct][j + 8 * h], s1[j]);
      atomicAdd(&rssL[ct][j + 8 * h], s2[j]);
    }
  }
  __syncthreads();

  float mu[8], rstd[8];
  #pragma unroll
  for (int j = 0; j < 8; ++j) {
    const float m_ = rsL[ct][j + 8 * h] * (1.0f / (float)D_);
    const float v_ = rssL[ct][j + 8 * h] * (1.0f / (float)D_) - m_ * m_;
    mu[j] = m_;
    rstd[j] = rsqrtf(v_ + LN_EPS);
  }
  float po[8];
  #pragma unroll
  for (int j = 0; j < 8; ++j) po[j] = 0.f;
  #pragma unroll
  for (int t = 0; t < 12; ++t) {
    const int d = dbase + t * 16 + ln;
    const float g = gamma[d], be = beta[d], ow = out_w[d];
    #pragma unroll
    for (int j = 0; j < 8; ++j)
      po[j] += ((acc[t][j] - mu[j]) * rstd[j] * g + be) * ow;
  }
  #pragma unroll
  for (int j = 0; j < 8; ++j) {
    #pragma unroll
    for (int msk = 1; msk <= 8; msk <<= 1)
      po[j] += __shfl_xor(po[j], msk, 32);
  }
  if (ln == 0) {
    #pragma unroll
    for (int j = 0; j < 8; ++j) atomicAdd(&orL[ct][j + 8 * h], po[j]);
  }
  __syncthreads();

  if (tid < 32) {
    const int cti = tid >> 4, rr = tid & 15;
    out[(size_t)b * C_ + blockIdx.x * 32 + cti * 16 + rr] =
        orL[cti][rr] + out_b[0];
  }
}

// ------------------------------------------------------------------ launch ---
extern "C" void kernel_launch(void* const* d_in, const int* in_sizes, int n_in,
                              void* d_out, int out_size, void* d_ws, size_t ws_size,
                              hipStream_t stream) {
  (void)in_sizes; (void)n_in; (void)out_size; (void)ws_size;
  const float* x     = (const float*)d_in[0];
  const float* q     = (const float*)d_in[1];
  const float* Wk    = (const float*)d_in[2];
  const float* Wv    = (const float*)d_in[3];
  const float* bv    = (const float*)d_in[4];
  const float* gam   = (const float*)d_in[5];
  const float* bet   = (const float*)d_in[6];
  const float* ow    = (const float*)d_in[7];
  const float* ob    = (const float*)d_in[8];

  char* ws = (char*)d_ws;
  size_t off = 0;
  auto alloc = [&](size_t bytes) -> void* {
    void* pp = ws + off;
    off = (off + bytes + 255) & ~(size_t)255;
    return pp;
  };
  unsigned short* xbf  = (unsigned short*)alloc((size_t)B_ * S_ * D_ * 2);
  unsigned short* qbf  = (unsigned short*)alloc((size_t)C_ * D_ * 2);
  unsigned short* wkbf = (unsigned short*)alloc((size_t)D_ * D_ * 2);
  unsigned short* wvbf = (unsigned short*)alloc((size_t)D_ * D_ * 2);
  unsigned short* kbf  = (unsigned short*)alloc((size_t)B_ * S_ * D_ * 2);
  unsigned short* vtbf = (unsigned short*)alloc((size_t)B_ * D_ * S_ * 2);

  const int nx = B_ * S_ * D_, nq = C_ * D_, nw = D_ * D_;
  cvt_f32_bf16<<<(nx + 255) / 256, 256, 0, stream>>>(x, xbf, nx);
  cvt_f32_bf16<<<(nq + 255) / 256, 256, 0, stream>>>(q, qbf, nq);
  cvt_f32_bf16<<<(nw + 255) / 256, 256, 0, stream>>>(Wk, wkbf, nw);
  cvt_f32_bf16<<<(nw + 255) / 256, 256, 0, stream>>>(Wv, wvbf, nw);

  kv_gemm<<<dim3(S_ / 16, 6, 2 * B_), 256, 0, stream>>>(xbf, wkbf, wvbf, bv,
                                                        kbf, vtbf);
  attn<<<dim3(C_ / 32, B_), 256, 0, stream>>>(qbf, kbf, vtbf, gam, bet, ow, ob,
                                              (float*)d_out);
}
